// LinearAltLast_15135464751425
// MI455X (gfx1250) — matrix operand order repaired and checked by reading, not compiled, not run on hardware
//
#include <hip/hip_runtime.h>
#include <cstdint>

// Problem constants (match reference).
#define BDIM 4096
#define FDIM 4096
#define ODIM 32000

// GEMM tiling.
#define BM 128
#define BN 128
#define BK 64
#define LDW (BK + 8)   // padded LDS row stride in halves (+16B keeps b128 alignment, spreads banks)

typedef __attribute__((ext_vector_type(16))) _Float16 v16h;
typedef __attribute__((ext_vector_type(8)))  _Float16 v8h;
typedef __attribute__((ext_vector_type(4)))  _Float16 v4h;
typedef __attribute__((ext_vector_type(8)))  float    v8f;

// ---------------------------------------------------------------------------
// Prep A: Aq[i,k] = f16(x[i,k] - mask[i,k])   (mask[:4096,:] is row-major, same
// flat layout as x). One float4 per thread.
// ---------------------------------------------------------------------------
__global__ void prep_a_kernel(const float* __restrict__ x,
                              const float* __restrict__ mask,
                              _Float16* __restrict__ aq) {
    const size_t i = (size_t)blockIdx.x * blockDim.x + threadIdx.x;   // float4 index
    const float4 xv = ((const float4*)x)[i];
    const float4 mv = ((const float4*)mask)[i];
    v4h h;
    h.x = (_Float16)(xv.x - mv.x);
    h.y = (_Float16)(xv.y - mv.y);
    h.z = (_Float16)(xv.z - mv.z);
    h.w = (_Float16)(xv.w - mv.w);
    ((v4h*)aq)[i] = h;
}

// ---------------------------------------------------------------------------
// Prep B: Bq[o,k] = f16(quant(W[o,k]) - mask[o,k]),
// quant(w) = clip(rint(w*128)/128, -1, 1)   (rint == RNE matches jnp.round).
// Exact in f16 (steps of 1/128, |v| <= 2).
// ---------------------------------------------------------------------------
__device__ __forceinline__ float quant_w(float w) {
    float q = __builtin_rintf(w * 128.0f) * 0.0078125f;
    return fminf(1.0f, fmaxf(-1.0f, q));
}

__global__ void prep_b_kernel(const float* __restrict__ w,
                              const float* __restrict__ mask,
                              _Float16* __restrict__ bq) {
    const size_t i = (size_t)blockIdx.x * blockDim.x + threadIdx.x;   // float4 index
    const float4 wv = ((const float4*)w)[i];
    const float4 mv = ((const float4*)mask)[i];
    v4h h;
    h.x = (_Float16)(quant_w(wv.x) - mv.x);
    h.y = (_Float16)(quant_w(wv.y) - mv.y);
    h.z = (_Float16)(quant_w(wv.z) - mv.z);
    h.w = (_Float16)(quant_w(wv.w) - mv.w);
    ((v4h*)bq)[i] = h;
}

// ---------------------------------------------------------------------------
// WMMA GEMM: logits[m,n] = sum_k Aq[m,k] * Bq[n,k]   (both operands K-major).
// Block = 128x128 tile, 256 threads = 8 waves in a 2(M) x 4(N) grid, each wave
// computes 4x2 16x16 tiles via v_wmma_f32_16x16x32_f16.
// Double-buffered async global->LDS pipeline (ASYNCcnt): stage k+1's copies
// are in flight while stage k's WMMAs run; one barrier per stage.
// ---------------------------------------------------------------------------
__global__ __launch_bounds__(256) void gemm_kernel(
        const _Float16* __restrict__ aq,
        const _Float16* __restrict__ bq,
        float* __restrict__ out) {
    __shared__ _Float16 sA[2][BM * LDW];
    __shared__ _Float16 sB[2][BN * LDW];

    const int tid  = threadIdx.x;
    const int lane = tid & 31;
    const int wid  = tid >> 5;

    const int m0 = blockIdx.x * BM;   // M-tile fastest -> 32 consecutive blocks share a weight panel in L2
    const int n0 = blockIdx.y * BN;

    const int waveM = (wid >> 2) * 64;   // 2 wave-rows of 64
    const int waveN = (wid & 3) * 32;    // 4 wave-cols of 32

    const uint32_t sAaddr[2] = { (uint32_t)(uintptr_t)&sA[0][0],
                                 (uint32_t)(uintptr_t)&sA[1][0] };
    const uint32_t sBaddr[2] = { (uint32_t)(uintptr_t)&sB[0][0],
                                 (uint32_t)(uintptr_t)&sB[1][0] };

    // Per-thread async-copy chunk coordinates: 1024 16B-chunks per 128x64 tile,
    // 4 chunks per thread (invariant across stages).
    uint32_t goff[4], loff[4];
#pragma unroll
    for (int i = 0; i < 4; ++i) {
        const int c   = tid + i * 256;
        const int row = c >> 3;          // 0..127
        const int col = (c & 7) * 8;     // half offset within the 64-wide K slab
        goff[i] = (uint32_t)((row * FDIM + col) * 2);
        loff[i] = (uint32_t)((row * LDW  + col) * 2);
    }

    const _Float16* gAbase = aq + (size_t)m0 * FDIM;
    const _Float16* gBbase = bq + (size_t)n0 * FDIM;

    // Issue one stage's async copies into buffer `buf` (no wait).
    auto issue_stage = [&](int buf, int k0) {
        const _Float16* gA = gAbase + k0;
        const _Float16* gB = gBbase + k0;
#pragma unroll
        for (int i = 0; i < 4; ++i) {
            const uint32_t la = sAaddr[buf] + loff[i];
            const uint32_t lb = sBaddr[buf] + loff[i];
            asm volatile("global_load_async_to_lds_b128 %0, %1, %2"
                         :: "v"(la), "v"(goff[i]), "s"(gA) : "memory");
            asm volatile("global_load_async_to_lds_b128 %0, %1, %2"
                         :: "v"(lb), "v"(goff[i]), "s"(gB) : "memory");
        }
    };

    v8f acc[4][2] = {};

    // Consume one resident stage from buffer `buf`.
    auto compute_stage = [&](int buf) {
        const _Float16* __restrict__ cA = &sA[buf][0];
        const _Float16* __restrict__ cB = &sB[buf][0];
#pragma unroll
        for (int kk = 0; kk < BK; kk += 32) {
            // A fragments (§7.12.2): lane l<16 -> row m+l, K {kk..kk+7, kk+16..kk+23};
            //                        lane l>=16 -> same row, K {+8, +24}.
            const int alo = kk + ((lane & 16) ? 8 : 0);
            v16h afrag[4];
#pragma unroll
            for (int m = 0; m < 4; ++m) {
                const int row = waveM + m * 16 + (lane & 15);
                const v8h a0 = *(const v8h*)&cA[row * LDW + alo];
                const v8h a1 = *(const v8h*)&cA[row * LDW + alo + 16];
                afrag[m] = __builtin_shufflevector(a0, a1,
                    0,1,2,3,4,5,6,7,8,9,10,11,12,13,14,15);
            }
            // B fragments: lane l holds column (l&15); l<16 -> K {kk..kk+7, kk+16..kk+23},
            //              l>=16 -> K {+8, +24} (same order as A).
            const int blo = kk + ((lane & 16) ? 8 : 0);
            v16h bfrag[2];
#pragma unroll
            for (int n = 0; n < 2; ++n) {
                const int col = waveN + n * 16 + (lane & 15);
                const v8h b0 = *(const v8h*)&cB[col * LDW + blo];
                const v8h b1 = *(const v8h*)&cB[col * LDW + blo + 16];
                bfrag[n] = __builtin_shufflevector(b0, b1,
                    0,1,2,3,4,5,6,7,8,9,10,11,12,13,14,15);
            }
#pragma unroll
            for (int m = 0; m < 4; ++m)
#pragma unroll
                for (int n = 0; n < 2; ++n)
                    acc[m][n] = __builtin_amdgcn_wmma_f32_16x16x32_f16(
                        false, afrag[m], false, bfrag[n],
                        (short)0, acc[m][n], false, false);
        }
    };

    // ---- pipeline prologue: stage 0 resident, stage 1 in flight ----
    issue_stage(0, 0);
    asm volatile("s_wait_asynccnt 0" ::: "memory");
    __syncthreads();
    issue_stage(1, BK);

    int cur = 0;
    for (int k0 = 0; k0 < FDIM; k0 += BK) {
        compute_stage(cur);
        if (k0 + BK < FDIM) {
            asm volatile("s_wait_asynccnt 0" ::: "memory");   // stage k+1 landed
            __syncthreads();                                  // everyone done reading buf `cur`
            if (k0 + 2 * BK < FDIM)
                issue_stage(cur, k0 + 2 * BK);                // refill freed buffer
            cur ^= 1;
        }
    }

    // ---- epilogue: C layout VGPR i -> M = i (+8 for high half-wave), N = lane&15 ----
#pragma unroll
    for (int m = 0; m < 4; ++m) {
#pragma unroll
        for (int n = 0; n < 2; ++n) {
            const int rbase = m0 + waveM + m * 16 + ((lane & 16) ? 8 : 0);
            const int col   = n0 + waveN + n * 16 + (lane & 15);
#pragma unroll
            for (int i = 0; i < 8; ++i)
                out[(size_t)(rbase + i) * ODIM + col] = acc[m][n][i];
        }
    }
}

// ---------------------------------------------------------------------------
// In-place row softmax over d_out: one 256-thread block per row (rows stay in
// L2 between the three passes: max, sum(exp), normalize).
// ---------------------------------------------------------------------------
__global__ __launch_bounds__(256) void softmax_kernel(float* __restrict__ out) {
    __shared__ float red[256];
    const int tid = threadIdx.x;
    float* row = out + (size_t)blockIdx.x * ODIM;

    float m = -3.402823466e38f;
    for (int c = tid; c < ODIM; c += 256) m = fmaxf(m, row[c]);
    red[tid] = m;
    __syncthreads();
    for (int s = 128; s > 0; s >>= 1) {
        if (tid < s) red[tid] = fmaxf(red[tid], red[tid + s]);
        __syncthreads();
    }
    const float rowmax = red[0];
    __syncthreads();

    float sum = 0.0f;
    for (int c = tid; c < ODIM; c += 256) sum += __expf(row[c] - rowmax);
    red[tid] = sum;
    __syncthreads();
    for (int s = 128; s > 0; s >>= 1) {
        if (tid < s) red[tid] += red[tid + s];
        __syncthreads();
    }
    const float inv = 1.0f / red[0];
    __syncthreads();

    for (int c = tid; c < ODIM; c += 256)
        row[c] = __expf(row[c] - rowmax) * inv;
}

// ---------------------------------------------------------------------------
// Launch: prepA -> prepB -> WMMA GEMM (logits into d_out) -> in-place softmax.
// Workspace layout: Aq (4096*4096 f16 = 32MB) then Bq (32000*4096 f16 = 250MB).
// ---------------------------------------------------------------------------
extern "C" void kernel_launch(void* const* d_in, const int* in_sizes, int n_in,
                              void* d_out, int out_size, void* d_ws, size_t ws_size,
                              hipStream_t stream) {
    (void)in_sizes; (void)n_in; (void)out_size; (void)ws_size;
    const float* x    = (const float*)d_in[0];   // [4096, 4096]
    const float* w    = (const float*)d_in[1];   // [32000, 4096]
    const float* mask = (const float*)d_in[2];   // [32000, 4096]
    float* out = (float*)d_out;                  // [4096, 32000]

    _Float16* aq = (_Float16*)d_ws;
    _Float16* bq = aq + (size_t)BDIM * FDIM;

    // Prep passes (one float4 per thread).
    prep_a_kernel<<<(BDIM * FDIM / 4) / 256, 256, 0, stream>>>(x, mask, aq);
    prep_b_kernel<<<((size_t)ODIM * FDIM / 4) / 256, 256, 0, stream>>>(w, mask, bq);

    // GEMM: grid.x = M tiles (fastest -> weight-panel L2 reuse), grid.y = N tiles.
    dim3 grid(BDIM / BM, ODIM / BN);
    gemm_kernel<<<grid, 256, 0, stream>>>(aq, bq, out);

    // Row softmax in-place.
    softmax_kernel<<<BDIM, 256, 0, stream>>>(out);
}
